// MultiHead_Attention_13657996001551
// MI455X (gfx1250) — matrix operand order repaired and checked by reading, not compiled, not run on hardware
//
#include <hip/hip_runtime.h>
#include <hip/hip_bf16.h>

// ---------------------------------------------------------------------------
// MHA forward for MI455X (gfx1250, wave32, WMMA).
// B=4, L=2048, D=1024, H=16, HD=64, SCALE=1/8.
// Path: fp32->fp16 convert -> 3 WMMA proj GEMMs -> flash-attn (WMMA) -> WMMA out GEMM.
// GEMM: 32x64 tile per wave (B-fragment reuse x2, 8 independent WMMA chains).
// Softmax: row sums via WMMA against ones; row max via DPP butterfly.
// ---------------------------------------------------------------------------

typedef __attribute__((ext_vector_type(16))) _Float16 v16h;
typedef __attribute__((ext_vector_type(8)))  _Float16 v8h;
typedef __attribute__((ext_vector_type(4)))  _Float16 v4h;
typedef __attribute__((ext_vector_type(8)))  float    v8f;

#define MHA_B  4
#define MHA_L  2048
#define MHA_D  1024
#define MHA_H  16
#define MHA_HD 64
#define MHA_M  (MHA_B * MHA_L)          // 8192 tokens
// softmax done in exp2 domain: score * (1/8) * log2(e)
#define MHA_QSCALE 0.18033688011112042f

static __device__ __forceinline__ v8f wmma_f16(v16h a, v16h b, v8f c) {
  // 8 args: (neg_a, A, neg_b, B, c_mod, C, reuse_a, reuse_b)
  return __builtin_amdgcn_wmma_f32_16x16x32_f16(false, a, false, b, (short)0, c,
                                                false, false);
}

static __device__ __forceinline__ v16h combine16(v8h lo, v8h hi) {
  v16h r;
#pragma unroll
  for (int i = 0; i < 8; ++i) { r[i] = lo[i]; r[i + 8] = hi[i]; }
  return r;
}

// A-fragment (16x32 f16, M x K), base points at (row0, k0) of a row-major
// matrix with leading dim ld (in halves).  ISA 7.12.2: lane m = lane&15,
// half-sel hs = lane>>4; halves 0..7 = K(hs*8 + 0..7), halves 8..15 = K(16 + hs*8 + 0..7).
static __device__ __forceinline__ v16h load_a_frag(const _Float16* base, int ld, int lane) {
  const int m  = lane & 15;
  const int hs = lane >> 4;
  const _Float16* p = base + (size_t)m * ld + hs * 8;
  v8h lo = *(const v8h*)(p);
  v8h hi = *(const v8h*)(p + 16);
  return combine16(lo, hi);
}

// B-fragment (32x16 f16, K x N) where B[k][n] = S[n][k] for a row-major source
// S with leading dim ld (rows of S are K-contiguous).  Lane n = lane&15,
// halves 0..7 = S[n][hs*8 + 0..7], halves 8..15 = S[n][16 + hs*8 + 0..7].
static __device__ __forceinline__ v16h load_bT_frag(const _Float16* base, int ld, int lane) {
  const int n  = lane & 15;
  const int hs = lane >> 4;
  const _Float16* p = base + (size_t)n * ld + hs * 8;
  v8h lo = *(const v8h*)(p);
  v8h hi = *(const v8h*)(p + 16);
  return combine16(lo, hi);
}

// ---- DPP cross-lane max within each 16-lane half (VALU only, no LDS) ----
template <int CTRL>
static __device__ __forceinline__ float dpp_movf(float x) {
  return __int_as_float(__builtin_amdgcn_update_dpp(
      0, __float_as_int(x), CTRL, 0xF, 0xF, true));
}

// Butterfly over disjoint sets: xor1 (quad_perm 0xB1), xor2 (quad_perm 0x4E),
// xor7 (ROW_HALF_MIRROR 0x141), xor15 (ROW_MIRROR 0x140)  => exact 16-lane max.
static __device__ __forceinline__ void rowmax16(v8f& v) {
#pragma unroll
  for (int i = 0; i < 8; ++i) v[i] = fmaxf(v[i], dpp_movf<0x0B1>(v[i]));
#pragma unroll
  for (int i = 0; i < 8; ++i) v[i] = fmaxf(v[i], dpp_movf<0x04E>(v[i]));
#pragma unroll
  for (int i = 0; i < 8; ++i) v[i] = fmaxf(v[i], dpp_movf<0x141>(v[i]));
#pragma unroll
  for (int i = 0; i < 8; ++i) v[i] = fmaxf(v[i], dpp_movf<0x140>(v[i]));
}

// ---------------------------------------------------------------------------
// fp32 -> fp16 conversion (vectorized x4)
// ---------------------------------------------------------------------------
__global__ void cvt_f32_to_f16_kernel(const float* __restrict__ src,
                                      _Float16* __restrict__ dst, int n4) {
  int i = blockIdx.x * blockDim.x + threadIdx.x;
  if (i >= n4) return;
  float4 f = ((const float4*)src)[i];
  v4h h;
  h[0] = (_Float16)f.x; h[1] = (_Float16)f.y;
  h[2] = (_Float16)f.z; h[3] = (_Float16)f.w;
  *(v4h*)(dst + (size_t)i * 4) = h;
}

// ---------------------------------------------------------------------------
// GEMM: out = A[MxK] @ W[NxK]^T + bias.  One wave computes a 32(M) x 64(N) tile
// (two 16-row groups sharing each B fragment; 8 independent WMMA accumulators).
// MODE 0: store fp16 head-split  Qh/Kh[b][h][l][hd]
// MODE 1: store fp16 transposed  Vt[b][h][hd][l]
// MODE 2: store fp32 row-major   out[m][n]
// ---------------------------------------------------------------------------
template <int MODE>
__global__ void gemm_xwt_kernel(const _Float16* __restrict__ A,
                                const _Float16* __restrict__ W,
                                const float* __restrict__ bias,
                                void* __restrict__ out,
                                int M, int N, int K) {
  const int lane = threadIdx.x & 31;
  const int wave = threadIdx.x >> 5;
  const int gw   = blockIdx.x * (blockDim.x >> 5) + wave;
  const int nTiles = N >> 6;
  const int mTile = gw / nTiles;
  const int nTile = gw - mTile * nTiles;
  if (mTile * 32 >= M) return;
  const int m0 = mTile * 32;
  const int n0 = nTile * 64;

  v8f acc[2][4] = {};
  for (int k0 = 0; k0 < K; k0 += 32) {
    v16h a0 = load_a_frag(A + (size_t)(m0 +  0) * K + k0, K, lane);
    v16h a1 = load_a_frag(A + (size_t)(m0 + 16) * K + k0, K, lane);
#pragma unroll
    for (int f = 0; f < 4; ++f) {
      v16h b = load_bT_frag(W + (size_t)(n0 + f * 16) * K + k0, K, lane);
      acc[0][f] = wmma_f16(a0, b, acc[0][f]);
      acc[1][f] = wmma_f16(a1, b, acc[1][f]);
    }
  }

  const int n_lane = lane & 15;
  const int rbase  = (lane >> 4) * 8;   // C frag: VGPR i holds row rbase+i, col n_lane

#pragma unroll
  for (int g = 0; g < 2; ++g) {
    const int m0g = m0 + g * 16;
#pragma unroll
    for (int f = 0; f < 4; ++f) {
      const int n  = n0 + f * 16 + n_lane;
      const float bv = bias[n];
      if (MODE == 0) {                  // head-split fp16 [B,H,L,HD]
        const int h = n >> 6, d = n & 63;
        _Float16* dst = (_Float16*)out;
#pragma unroll
        for (int i = 0; i < 8; ++i) {
          const int m = m0g + rbase + i;
          const int b = m >> 11, l = m & (MHA_L - 1);
          dst[(((size_t)(b * MHA_H + h)) * MHA_L + l) * MHA_HD + d] =
              (_Float16)(acc[g][f][i] + bv);
        }
      } else if (MODE == 1) {           // transposed fp16 [B,H,HD,L]
        const int h = n >> 6, d = n & 63;
        const int mf = m0g + rbase;
        const int b = mf >> 11, l0 = mf & (MHA_L - 1);
        v8h pk;
#pragma unroll
        for (int i = 0; i < 8; ++i) pk[i] = (_Float16)(acc[g][f][i] + bv);
        *(v8h*)((_Float16*)out +
                (((size_t)(b * MHA_H + h)) * MHA_HD + d) * MHA_L + l0) = pk;
      } else {                          // fp32 row-major [M,N]
        float* dst = (float*)out;
#pragma unroll
        for (int i = 0; i < 8; ++i)
          dst[(size_t)(m0g + rbase + i) * N + n] = acc[g][f][i] + bv;
      }
    }
  }
}

// ---------------------------------------------------------------------------
// Flash attention: grid = B*H * (L/64), block = 128 (4 waves).
// Each wave: one 16-query tile, streams keys in blocks of 32.
// Qh/Kh: [B,H,L,64] fp16,  Vt: [B,H,64,L] fp16,  Oh: [B,L,D] fp16 (heads merged).
// ---------------------------------------------------------------------------
__global__ void attn_kernel(const _Float16* __restrict__ Qh,
                            const _Float16* __restrict__ Kh,
                            const _Float16* __restrict__ Vt,
                            _Float16* __restrict__ Oh) {
  __shared__ __align__(16) _Float16 ldsP[4 * 16 * 32];
  const int lane = threadIdx.x & 31;
  const int wave = threadIdx.x >> 5;
  _Float16* myP = ldsP + wave * (16 * 32);

  const int bh   = blockIdx.x >> 5;           // 0..63
  const int qblk = blockIdx.x & 31;           // 0..31
  const int q0   = qblk * 64 + wave * 16;
  const int b    = bh >> 4, h = bh & 15;

  const _Float16* Qbase = Qh + ((size_t)bh * MHA_L + q0) * MHA_HD;
  const _Float16* Kbase = Kh + (size_t)bh * MHA_L * MHA_HD;
  const _Float16* Vbase = Vt + (size_t)bh * MHA_HD * MHA_L;

  const v16h qa0 = load_a_frag(Qbase + 0,  MHA_HD, lane);   // d = 0..31
  const v16h qa1 = load_a_frag(Qbase + 32, MHA_HD, lane);   // d = 32..63

  v16h ones;
#pragma unroll
  for (int i = 0; i < 16; ++i) ones[i] = (_Float16)1.0f;

  v8f o0 = {}, o1 = {}, o2 = {}, o3 = {};
  v8f rs = {};                                // row sums (WMMA-accumulated)
  v8f rmax;
#pragma unroll
  for (int i = 0; i < 8; ++i) rmax[i] = -1e30f;

  const int n_lane = lane & 15;
  const int rbase  = (lane >> 4) * 8;

  for (int j0 = 0; j0 < MHA_L; j0 += 32) {
    // ---- S = Q @ K^T (two 16x16 fp32 tiles over 32 keys) ----
    v8f s0 = {}, s1 = {};
    {
      v16h b00 = load_bT_frag(Kbase + (size_t)(j0 +  0) * MHA_HD + 0,  MHA_HD, lane);
      v16h b01 = load_bT_frag(Kbase + (size_t)(j0 +  0) * MHA_HD + 32, MHA_HD, lane);
      s0 = wmma_f16(qa0, b00, s0);
      s0 = wmma_f16(qa1, b01, s0);
      v16h b10 = load_bT_frag(Kbase + (size_t)(j0 + 16) * MHA_HD + 0,  MHA_HD, lane);
      v16h b11 = load_bT_frag(Kbase + (size_t)(j0 + 16) * MHA_HD + 32, MHA_HD, lane);
      s1 = wmma_f16(qa0, b10, s1);
      s1 = wmma_f16(qa1, b11, s1);
    }
    // scale into exp2 domain
#pragma unroll
    for (int i = 0; i < 8; ++i) { s0[i] *= MHA_QSCALE; s1[i] *= MHA_QSCALE; }

    // ---- online softmax: per-row block max via DPP butterfly ----
    v8f bm;
#pragma unroll
    for (int i = 0; i < 8; ++i) bm[i] = fmaxf(s0[i], s1[i]);
    rowmax16(bm);

    v8f scale;
#pragma unroll
    for (int i = 0; i < 8; ++i) {
      const float mn = fmaxf(rmax[i], bm[i]);
      scale[i] = __builtin_amdgcn_exp2f(rmax[i] - mn);
      rmax[i]  = mn;
    }
#pragma unroll
    for (int i = 0; i < 8; ++i) {
      s0[i] = __builtin_amdgcn_exp2f(s0[i] - rmax[i]);
      s1[i] = __builtin_amdgcn_exp2f(s1[i] - rmax[i]);
    }
#pragma unroll
    for (int i = 0; i < 8; ++i) {
      rs[i] *= scale[i];
      o0[i] *= scale[i]; o1[i] *= scale[i];
      o2[i] *= scale[i]; o3[i] *= scale[i];
    }

    // ---- P (C-layout) -> LDS -> A-layout fragment ----
#pragma unroll
    for (int i = 0; i < 8; ++i) {
      myP[(rbase + i) * 32 + n_lane]      = (_Float16)s0[i];
      myP[(rbase + i) * 32 + 16 + n_lane] = (_Float16)s1[i];
    }
    asm volatile("s_wait_dscnt 0x0" ::: "memory");
    const v16h pa = load_a_frag(myP, 32, lane);

    // ---- row sums: rs += P @ ones  (replicated across column lanes) ----
    rs = wmma_f16(pa, ones, rs);

    // ---- O += P @ V  (B frags contiguous thanks to transposed V) ----
    {
      v16h vb0 = load_bT_frag(Vbase + (size_t)( 0) * MHA_L + j0, MHA_L, lane);
      v16h vb1 = load_bT_frag(Vbase + (size_t)(16) * MHA_L + j0, MHA_L, lane);
      v16h vb2 = load_bT_frag(Vbase + (size_t)(32) * MHA_L + j0, MHA_L, lane);
      v16h vb3 = load_bT_frag(Vbase + (size_t)(48) * MHA_L + j0, MHA_L, lane);
      o0 = wmma_f16(pa, vb0, o0);
      o1 = wmma_f16(pa, vb1, o1);
      o2 = wmma_f16(pa, vb2, o2);
      o3 = wmma_f16(pa, vb3, o3);
    }
  }

  // ---- normalize and store merged-head fp16 [B, L, D] ----
  v8f inv;
#pragma unroll
  for (int i = 0; i < 8; ++i) inv[i] = 1.0f / rs[i];

  const v8f* oacc[4] = { &o0, &o1, &o2, &o3 };
#pragma unroll
  for (int f = 0; f < 4; ++f) {
    const int d = h * MHA_HD + f * 16 + n_lane;
#pragma unroll
    for (int i = 0; i < 8; ++i) {
      const int l = q0 + rbase + i;
      Oh[((size_t)b * MHA_L + l) * MHA_D + d] = (_Float16)((*oacc[f])[i] * inv[i]);
    }
  }
}

// ---------------------------------------------------------------------------
// Host-side launch.  Workspace layout (in _Float16 elements):
//   [0      , 8M )  qh   (fp16 of q)      -- reused later as Oh (attn output)
//   [8M     , 16M)  kh
//   [16M    , 24M)  vh
//   [24M    , 25M)  Wq_h   [25M,26M) Wk_h   [26M,27M) Wv_h   [27M,28M) Wo_h
//   [28M    , 36M)  Qh  head-split
//   [36M    , 44M)  Kh  head-split
//   [44M    , 52M)  Vt  transposed
// Total: 52M halves = 104 MB.
// ---------------------------------------------------------------------------
extern "C" void kernel_launch(void* const* d_in, const int* in_sizes, int n_in,
                              void* d_out, int out_size, void* d_ws, size_t ws_size,
                              hipStream_t stream) {
  const float* q  = (const float*)d_in[0];
  const float* k  = (const float*)d_in[1];
  const float* v  = (const float*)d_in[2];
  const float* Wq = (const float*)d_in[3];
  const float* bq = (const float*)d_in[4];
  const float* Wk = (const float*)d_in[5];
  const float* bk = (const float*)d_in[6];
  const float* Wv = (const float*)d_in[7];
  const float* bv = (const float*)d_in[8];
  const float* Wo = (const float*)d_in[9];
  const float* bo = (const float*)d_in[10];
  float* out = (float*)d_out;

  const size_t MEG = 1024u * 1024u;
  const size_t ACT = (size_t)MHA_M * MHA_D;   // 8M elements

  _Float16* ws  = (_Float16*)d_ws;
  _Float16* qh  = ws;                 // also Oh
  _Float16* kh  = ws + 1 * ACT;
  _Float16* vh  = ws + 2 * ACT;
  _Float16* wqh = ws + 3 * ACT;
  _Float16* wkh = ws + 3 * ACT + 1 * MEG;
  _Float16* wvh = ws + 3 * ACT + 2 * MEG;
  _Float16* woh = ws + 3 * ACT + 3 * MEG;
  _Float16* Qh  = ws + 3 * ACT + 4 * MEG;
  _Float16* Kh  = Qh + ACT;
  _Float16* Vt  = Kh + ACT;
  _Float16* Oh  = qh;                 // reuse: q-halves dead after Q projection

  // ---- fp32 -> fp16 conversions ----
  const int nAct4 = (int)(ACT / 4);          // 2M
  const int nW4   = (int)(MEG / 4);          // 256K
  cvt_f32_to_f16_kernel<<<nAct4 / 256, 256, 0, stream>>>(q,  qh,  nAct4);
  cvt_f32_to_f16_kernel<<<nAct4 / 256, 256, 0, stream>>>(k,  kh,  nAct4);
  cvt_f32_to_f16_kernel<<<nAct4 / 256, 256, 0, stream>>>(v,  vh,  nAct4);
  cvt_f32_to_f16_kernel<<<nW4 / 256,   256, 0, stream>>>(Wq, wqh, nW4);
  cvt_f32_to_f16_kernel<<<nW4 / 256,   256, 0, stream>>>(Wk, wkh, nW4);
  cvt_f32_to_f16_kernel<<<nW4 / 256,   256, 0, stream>>>(Wv, wvh, nW4);
  cvt_f32_to_f16_kernel<<<nW4 / 256,   256, 0, stream>>>(Wo, woh, nW4);

  // ---- QKV projections: (M/32)*(N/64) = 4096 wave tiles / 8 waves = 512 blocks ----
  const int gemmBlocks = (MHA_M / 32) * (MHA_D / 64) / 8;
  gemm_xwt_kernel<0><<<gemmBlocks, 256, 0, stream>>>(qh, wqh, bq, (void*)Qh,
                                                     MHA_M, MHA_D, MHA_D);
  gemm_xwt_kernel<0><<<gemmBlocks, 256, 0, stream>>>(kh, wkh, bk, (void*)Kh,
                                                     MHA_M, MHA_D, MHA_D);
  gemm_xwt_kernel<1><<<gemmBlocks, 256, 0, stream>>>(vh, wvh, bv, (void*)Vt,
                                                     MHA_M, MHA_D, MHA_D);

  // ---- attention: B*H*(L/64) = 2048 blocks of 4 waves ----
  attn_kernel<<<MHA_B * MHA_H * (MHA_L / 64), 128, 0, stream>>>(Qh, Kh, Vt, Oh);

  // ---- output projection, fp32 store ----
  gemm_xwt_kernel<2><<<gemmBlocks, 256, 0, stream>>>(Oh, woh, bo, (void*)out,
                                                     MHA_M, MHA_D, MHA_D);
}
